// TimeSpaceAttention_11218454577592
// MI455X (gfx1250) — matrix operand order repaired and checked by reading, compiled, not run on hardware
//
#include <hip/hip_runtime.h>
#include <hip/hip_bf16.h>

// ---------------- problem constants ----------------
#define E_DIM   768
#define H_HEADS 12
#define T_DIM   8
#define S_DIM   196
#define L_DIM   1569      // 1 + S*T
#define B_DIM   8
#define HD      64        // head dim
#define BH      96        // B * H
#define SP      197       // spatial slots (cls + S)
#define M_VALID 12552     // L * B
#define M_PAD   12672     // 99 * 128  (GEMM row padding)
#define SP_PAD  208       // 13 * 16   (query/key row padding)
#define ZP      224       // 7 * 32    (z padded to WMMA K granularity)

typedef __attribute__((ext_vector_type(16))) _Float16 v16h;
typedef __attribute__((ext_vector_type(8)))  _Float16 v8h;
typedef __attribute__((ext_vector_type(8)))  float    v8f;
typedef __attribute__((ext_vector_type(4)))  int      v4i;

#if __has_builtin(__builtin_amdgcn_global_load_async_to_lds_b128)
#define HAVE_ASYNC_LDS 1
#endif

// async global(16B/lane) -> LDS copy; falls back to sync VGPR round-trip.
static __device__ __forceinline__ void async_copy16(const _Float16* g,
                                                    _Float16* l) {
#ifdef HAVE_ASYNC_LDS
  __builtin_amdgcn_global_load_async_to_lds_b128(
      (__attribute__((address_space(1))) v4i*)(v4i*)(void*)g,
      (__attribute__((address_space(3))) v4i*)(v4i*)(void*)l, 0, 0);
#else
  *(v8h*)l = *(const v8h*)g;
#endif
}

static __device__ __forceinline__ void wait_async0() {
#if __has_builtin(__builtin_amdgcn_s_wait_asynccnt)
  __builtin_amdgcn_s_wait_asynccnt(0);
#else
  asm volatile("s_wait_asynccnt 0x0" ::: "memory");
#endif
}

static __device__ __forceinline__ v8f vzero8f() {
  v8f z;
#pragma unroll
  for (int i = 0; i < 8; i++) z[i] = 0.f;
  return z;
}

static __device__ __forceinline__ v16h make_v16(v8h lo, v8h hi) {
  return __builtin_shufflevector(lo, hi, 0,1,2,3,4,5,6,7,8,9,10,11,12,13,14,15);
}

// A-matrix fragment (16x32 f16). Per CDNA5 ISA: lanes 0-15 hold row M=lane with
// K=0..7 (v0-3) and K=16..23 (v4-7); lanes 16-31 hold K=8..15 and K=24..31.
static __device__ __forceinline__ v16h load_a_frag(const _Float16* row, int kg) {
  v8h lo = *(const v8h*)(row + kg * 8);
  v8h hi = *(const v8h*)(row + 16 + kg * 8);
  return make_v16(lo, hi);
}

// B-matrix fragment (32x16 f16), stored as 16 rows of N with K contiguous
// (i.e. B^T row-major). lane: n = lane&15, kh = lane>>4: K=8kh..8kh+7, 16+8kh..23+8kh.
static __device__ __forceinline__ v16h load_b_frag(const _Float16* row, int kh) {
  v8h lo = *(const v8h*)(row + kh * 8);
  v8h hi = *(const v8h*)(row + 16 + kh * 8);
  return make_v16(lo, hi);
}

static __device__ __forceinline__ v8f wmma16(v16h a, v16h b, v8f c) {
  return __builtin_amdgcn_wmma_f32_16x16x32_f16(false, a, false, b, (short)0, c,
                                                false, false);
}

// ---------------- fp32 -> fp16 convert ----------------
__global__ void cvt_f32_f16(const float* __restrict__ in,
                            _Float16* __restrict__ out, int n) {
  int i = blockIdx.x * blockDim.x + threadIdx.x;
  if (i < n) out[i] = (_Float16)in[i];
}

// ---------------- WMMA GEMM:  C[M,N] = A[M,K] * W[N,K]^T + bias ----------------
// 128x128 block tile, BK=32, 8 waves, double-buffered async global->LDS stage.
__global__ __launch_bounds__(256) void gemm_wmma(
    const _Float16* __restrict__ A,   // M_PAD x K, row-major
    const _Float16* __restrict__ W,   // N x K, row-major
    const float* __restrict__ bias,   // N
    void* __restrict__ Cout, int K, int N, int mvalid, int out_f32) {
  __shared__ _Float16 As[2][128 * 40];   // stride 40 halves (80B, 16B aligned)
  __shared__ _Float16 Bs[2][128 * 40];

  const int tid  = threadIdx.x;
  const int lane = tid & 31;
  const int wave = tid >> 5;
  const int wm   = wave & 3;    // 4 M-quarters (32 rows each)
  const int wn   = wave >> 2;   // 2 N-halves  (64 cols each)
  const int nl   = lane & 15;
  const int half = lane >> 4;
  const long m0  = (long)blockIdx.x * 128;
  const int  n0  = blockIdx.y * 128;

  // issue one 128x32 A tile + 128x32 B tile as async 16B/lane copies
  auto issue_tile = [&](int buf, int k0) {
    for (int idx = tid; idx < 128 * 4; idx += 256) {
      int r = idx >> 2, c = idx & 3;
      async_copy16(&A[(m0 + r) * K + k0 + c * 8], &As[buf][r * 40 + c * 8]);
      async_copy16(&W[(long)(n0 + r) * K + k0 + c * 8], &Bs[buf][r * 40 + c * 8]);
    }
  };

  v8f acc[2][4];
#pragma unroll
  for (int i = 0; i < 2; i++)
#pragma unroll
    for (int j = 0; j < 4; j++) acc[i][j] = vzero8f();

  const int nk = K / 32;
  issue_tile(0, 0);
  for (int kt = 0; kt < nk; kt++) {
    const int buf = kt & 1;
    wait_async0();        // this wave's copies for tile kt have landed
    __syncthreads();      // ... and everyone else's
    if (kt + 1 < nk) issue_tile(buf ^ 1, (kt + 1) * 32);  // overlap with wmma

    v16h af[2], bf[4];
#pragma unroll
    for (int i = 0; i < 2; i++)
      af[i] = load_a_frag(&As[buf][(wm * 32 + i * 16 + nl) * 40], half);
#pragma unroll
    for (int j = 0; j < 4; j++)
      bf[j] = load_b_frag(&Bs[buf][(wn * 64 + j * 16 + nl) * 40], half);
#pragma unroll
    for (int i = 0; i < 2; i++)
#pragma unroll
      for (int j = 0; j < 4; j++) acc[i][j] = wmma16(af[i], bf[j], acc[i][j]);
  }

  // C frag layout: reg r -> row (r + 8*half), col = nl
#pragma unroll
  for (int i = 0; i < 2; i++)
#pragma unroll
    for (int j = 0; j < 4; j++) {
      int  n  = n0 + wn * 64 + j * 16 + nl;
      float bv = bias[n];
#pragma unroll
      for (int r = 0; r < 8; r++) {
        long  m = m0 + wm * 32 + i * 16 + half * 8 + r;
        float v = acc[i][j][r] + bv;
        if (out_f32) {
          if (m < mvalid) ((float*)Cout)[m * N + n] = v;
        } else {
          ((_Float16*)Cout)[m * N + n] = (_Float16)v;
        }
      }
    }
}

// ---------------- time-space attention core ----------------
// One block per (b = bh index, t). 8 waves; each wave owns 16-query-row tiles
// (13 tiles cover 197 spatial slots). Spatial logits + PV via WMMA, temporal
// (8-wide) part via VALU, masked softmax over the 205-way concat.
__global__ __launch_bounds__(256) void attn_kernel(
    const _Float16* __restrict__ Qb,  // [L][BH][HD] f16 (== GEMM output layout)
    const _Float16* __restrict__ Kb,
    const _Float16* __restrict__ Vb,
    float* __restrict__ Xsp,          // [BH][T][SP][HD] normalized spatial out
    float* __restrict__ Ptm)          // [BH][T][SP][T]  normalized temporal probs
{
  extern __shared__ __align__(16) char smem_raw[];
  _Float16* Qs = (_Float16*)smem_raw;          // SP_PAD x 72
  _Float16* Ks = Qs + SP_PAD * 72;             // SP_PAD x 72
  _Float16* Vt = Ks + SP_PAD * 72;             // HD x ZP (V transposed)
  _Float16* Ps = Vt + HD * ZP;                 // 8 waves x (16 x ZP) prob strips
  float*   TMs = (float*)(Ps + 8 * 16 * ZP);   // 8 waves x (16 x 8) tm logits

  const int b    = blockIdx.x;
  const int t    = blockIdx.y;
  const int bt   = b * T_DIM + t;
  const int tid  = threadIdx.x;
  const int lane = tid & 31;
  const int wave = tid >> 5;
  const int nl   = lane & 15;
  const int half = lane >> 4;

  v8h z8;
#pragma unroll
  for (int e = 0; e < 8; e++) z8[e] = (_Float16)0.f;

  // stage valid Q,K rows via async copies (row s -> token l; cls at s==0)
  for (int idx = tid; idx < SP * 8; idx += 256) {
    int s = idx >> 3, c = idx & 7;
    int  l   = (s == 0) ? 0 : 1 + (s - 1) * T_DIM + t;
    long off = ((long)l * BH + b) * HD + c * 8;
    async_copy16(&Qb[off], &Qs[s * 72 + c * 8]);
    async_copy16(&Kb[off], &Ks[s * 72 + c * 8]);
  }
  // zero pad rows (s = SP..SP_PAD-1)
  for (int idx = tid; idx < (SP_PAD - SP) * 8; idx += 256) {
    int s = SP + (idx >> 3), c = idx & 7;
    *(v8h*)&Qs[s * 72 + c * 8] = z8;
    *(v8h*)&Ks[s * 72 + c * 8] = z8;
  }
  // stage V transposed: Vt[d][z]
  for (int idx = tid; idx < HD * ZP; idx += 256) {
    int d = idx / ZP, z = idx % ZP;
    _Float16 val = (_Float16)0.f;
    if (z < SP) {
      int l = (z == 0) ? 0 : 1 + (z - 1) * T_DIM + t;
      val = Vb[((long)l * BH + b) * HD + d];
    }
    Vt[d * ZP + z] = val;
  }
  wait_async0();
  __syncthreads();

  _Float16* Pw = Ps + wave * 16 * ZP;
  float*    Tw = TMs + wave * 16 * 8;
  const float scale = 0.125f;  // 64^-0.5

  for (int tile = wave; tile < 13; tile += 8) {
    const int s0 = tile * 16;

    // ---- spatial logits: S[16 x 208] as 13 C-frags ----
    v8f sf[13];
#pragma unroll
    for (int j = 0; j < 13; j++) {
      v8f a = vzero8f();
#pragma unroll
      for (int kk = 0; kk < 2; kk++) {
        v16h af = load_a_frag(&Qs[(s0 + nl) * 72 + kk * 32], half);
        v16h bf = load_b_frag(&Ks[(j * 16 + nl) * 72 + kk * 32], half);
        a = wmma16(af, bf, a);
      }
      sf[j] = a;
    }

    // ---- temporal logits (VALU): lane -> (row = nl, u = half*4+uu) ----
    {
      int s = s0 + nl;
#pragma unroll
      for (int uu = 0; uu < 4; uu++) {
        int   u   = half * 4 + uu;
        float dot = -1e30f;
        if (s < SP) {
          int l = (s == 0) ? 0 : 1 + (s - 1) * T_DIM + u;
          const _Float16* kr = &Kb[((long)l * BH + b) * HD];
          const _Float16* qr = &Qs[s * 72];
          float a = 0.f;
#pragma unroll
          for (int c = 0; c < 8; c++) {
            v8h qv = *(const v8h*)(qr + c * 8);
            v8h kv = *(const v8h*)(kr + c * 8);
#pragma unroll
            for (int e = 0; e < 8; e++) a += (float)qv[e] * (float)kv[e];
          }
          dot = a * scale;
        }
        Tw[nl * 8 + u] = dot;
      }
    }

    // ---- masked softmax over [197 spatial + 8 temporal] per row ----
    float rinv[8];
#pragma unroll
    for (int r = 0; r < 8; r++) {
      const int row = r + half * 8;
      const int s   = s0 + row;
      float mx = -1e30f;
#pragma unroll
      for (int j = 0; j < 13; j++) {
        int   z = j * 16 + nl;
        float v = sf[j][r] * scale;
        sf[j][r] = v;
        if (z < SP) mx = fmaxf(mx, v);
      }
      mx = fmaxf(mx, __shfl_xor(mx, 1));
      mx = fmaxf(mx, __shfl_xor(mx, 2));
      mx = fmaxf(mx, __shfl_xor(mx, 4));
      mx = fmaxf(mx, __shfl_xor(mx, 8));
      float tl[8];
#pragma unroll
      for (int u = 0; u < 8; u++) {
        tl[u] = Tw[row * 8 + u];
        mx    = fmaxf(mx, tl[u]);
      }
      float sm = 0.f;
#pragma unroll
      for (int j = 0; j < 13; j++) {
        int   z = j * 16 + nl;
        float e = (z < SP) ? __expf(sf[j][r] - mx) : 0.f;
        sf[j][r] = e;
        sm += e;
      }
      sm += __shfl_xor(sm, 1);
      sm += __shfl_xor(sm, 2);
      sm += __shfl_xor(sm, 4);
      sm += __shfl_xor(sm, 8);
      float te[8];
#pragma unroll
      for (int u = 0; u < 8; u++) {
        te[u] = __expf(tl[u] - mx);
        sm += te[u];
      }
      float inv = 1.f / sm;
      rinv[r]   = inv;
      // normalized temporal probs: lane nl==u writes (rows r / r+8 per half)
#pragma unroll
      for (int u = 0; u < 8; u++)
        if (nl == u && s < SP) Ptm[((long)bt * SP + s) * 8 + u] = te[u] * inv;
      // unnormalized spatial probs -> per-wave LDS strip in A-frag order
#pragma unroll
      for (int j = 0; j < 13; j++)
        Pw[row * ZP + j * 16 + nl] = (_Float16)sf[j][r];
      Pw[row * ZP + 208 + nl] = (_Float16)0.f;  // K-pad to 224
    }

    // ---- PV: X[16 x 64] = P[16 x 224] * Vt^T ----
    v8f out[4];
#pragma unroll
    for (int dj = 0; dj < 4; dj++) out[dj] = vzero8f();
#pragma unroll
    for (int ks = 0; ks < 7; ks++) {
      v16h af = load_a_frag(&Pw[nl * ZP + ks * 32], half);
#pragma unroll
      for (int dj = 0; dj < 4; dj++) {
        v16h bf = load_b_frag(&Vt[(dj * 16 + nl) * ZP + ks * 32], half);
        out[dj] = wmma16(af, bf, out[dj]);
      }
    }
#pragma unroll
    for (int dj = 0; dj < 4; dj++)
#pragma unroll
      for (int r = 0; r < 8; r++) {
        int s = s0 + r + half * 8;
        if (s < SP)
          Xsp[((long)bt * SP + s) * HD + dj * 16 + nl] = out[dj][r] * rinv[r];
      }
  }
}

// ---------------- add temporal term, cls mean, scatter to xfull ----------------
__global__ void pack_kernel(const float* __restrict__ Xsp,
                            const float* __restrict__ Ptm,
                            const _Float16* __restrict__ Vb,
                            _Float16* __restrict__ xfull) {
  long idx = (long)blockIdx.x * blockDim.x + threadIdx.x;
  const long total = (long)BH * SP * T_DIM * HD;
  if (idx >= total) return;
  int  d = idx & 63;
  long r = idx >> 6;
  int  t = (int)(r & 7);
  r >>= 3;
  int s = (int)(r % SP);
  int b = (int)(r / SP);
  int batch = b / H_HEADS, h = b % H_HEADS;

  if (s == 0) {
    if (t != 0) return;  // one thread does the cls mean over t
    float accv = 0.f;
    float vv = (float)Vb[((long)0 * BH + b) * HD + d];  // cls V (l==0 for all u)
    for (int tt = 0; tt < T_DIM; tt++) {
      int bt = b * T_DIM + tt;
      float x = Xsp[((long)bt * SP + 0) * HD + d];
      const float* p = &Ptm[((long)bt * SP + 0) * 8];
      float psum = 0.f;
#pragma unroll
      for (int u = 0; u < 8; u++) psum += p[u];
      accv += x + psum * vv;
    }
    accv *= 0.125f;
    xfull[((long)0 * B_DIM + batch) * E_DIM + h * HD + d] = (_Float16)accv;
  } else {
    int bt = b * T_DIM + t;
    float x = Xsp[((long)bt * SP + s) * HD + d];
    const float* p = &Ptm[((long)bt * SP + s) * 8];
#pragma unroll
    for (int u = 0; u < 8; u++) {
      int lv = 1 + (s - 1) * T_DIM + u;
      x += p[u] * (float)Vb[((long)lv * BH + b) * HD + d];
    }
    int l = 1 + (s - 1) * T_DIM + t;
    xfull[((long)l * B_DIM + batch) * E_DIM + h * HD + d] = (_Float16)x;
  }
}

// ---------------- host orchestration ----------------
extern "C" void kernel_launch(void* const* d_in, const int* in_sizes, int n_in,
                              void* d_out, int out_size, void* d_ws,
                              size_t ws_size, hipStream_t stream) {
  (void)in_sizes; (void)n_in; (void)out_size; (void)ws_size;
  const float* q_in  = (const float*)d_in[0];
  const float* k_in  = (const float*)d_in[1];
  const float* v_in  = (const float*)d_in[2];
  const float* w_in  = (const float*)d_in[3];  // (2304, 768)
  const float* b_in  = (const float*)d_in[4];  // (2304,)
  const float* w_out = (const float*)d_in[5];  // (768, 768)
  const float* b_out = (const float*)d_in[6];  // (768,)
  float* out = (float*)d_out;

  char*  ws  = (char*)d_ws;
  size_t off = 0;
  auto alloc = [&](size_t bytes) -> char* {
    char* p = ws + off;
    off = (off + bytes + 255) & ~(size_t)255;
    return p;
  };

  _Float16* Xc   = (_Float16*)alloc((size_t)M_PAD * E_DIM * 2);  // conv input / xfull
  _Float16* Qb   = (_Float16*)alloc((size_t)M_PAD * E_DIM * 2);
  _Float16* Kb   = (_Float16*)alloc((size_t)M_PAD * E_DIM * 2);
  _Float16* Vb   = (_Float16*)alloc((size_t)M_PAD * E_DIM * 2);
  _Float16* W16  = (_Float16*)alloc((size_t)3 * E_DIM * E_DIM * 2);
  _Float16* Wo16 = (_Float16*)alloc((size_t)E_DIM * E_DIM * 2);
  float*    Xsp  = (float*)alloc((size_t)BH * T_DIM * SP * HD * 4);
  float*    Ptm  = (float*)alloc((size_t)BH * T_DIM * SP * 8 * 4);

  const int nX  = M_VALID * E_DIM;
  const int nW  = 3 * E_DIM * E_DIM;
  const int nWo = E_DIM * E_DIM;
  dim3 blk(256);
  dim3 gemm_grid(M_PAD / 128, E_DIM / 128);

  // weight conversion
  cvt_f32_f16<<<dim3((nW + 255) / 256), blk, 0, stream>>>(w_in, W16, nW);
  cvt_f32_f16<<<dim3((nWo + 255) / 256), blk, 0, stream>>>(w_out, Wo16, nWo);

  // QKV projections (stream-ordered; Xc reused per input)
  cvt_f32_f16<<<dim3((nX + 255) / 256), blk, 0, stream>>>(q_in, Xc, nX);
  gemm_wmma<<<gemm_grid, blk, 0, stream>>>(Xc, W16 + 0 * (size_t)E_DIM * E_DIM,
                                           b_in + 0, Qb, E_DIM, E_DIM, M_PAD, 0);
  cvt_f32_f16<<<dim3((nX + 255) / 256), blk, 0, stream>>>(k_in, Xc, nX);
  gemm_wmma<<<gemm_grid, blk, 0, stream>>>(Xc, W16 + 1 * (size_t)E_DIM * E_DIM,
                                           b_in + E_DIM, Kb, E_DIM, E_DIM, M_PAD, 0);
  cvt_f32_f16<<<dim3((nX + 255) / 256), blk, 0, stream>>>(v_in, Xc, nX);
  gemm_wmma<<<gemm_grid, blk, 0, stream>>>(Xc, W16 + 2 * (size_t)E_DIM * E_DIM,
                                           b_in + 2 * E_DIM, Vb, E_DIM, E_DIM, M_PAD, 0);

  // attention (dynamic LDS: Q + K + V^T + per-wave prob strips + tm logits)
  const int smem = (SP_PAD * 72 * 2) * 2 + HD * ZP * 2 + 8 * 16 * ZP * 2 +
                   8 * 16 * 8 * 4;  // = 150016 bytes (<320KB WGP LDS)
  attn_kernel<<<dim3(BH, T_DIM), blk, smem, stream>>>(Qb, Kb, Vb, Xsp, Ptm);

  // temporal add + cls mean -> xfull (reuses Xc)
  const long npack = (long)BH * SP * T_DIM * HD;
  pack_kernel<<<dim3((unsigned)((npack + 255) / 256)), blk, 0, stream>>>(
      Xsp, Ptm, Vb, Xc);

  // output projection -> fp32 d_out (bias added, M masked to 12552)
  gemm_wmma<<<gemm_grid, blk, 0, stream>>>(Xc, Wo16, b_out, out, E_DIM, E_DIM,
                                           M_VALID, 1);
}
